// MixtureOfExperts_15693810499844
// MI455X (gfx1250) — matrix operand order repaired and checked by reading, not compiled, not run on hardware
//
#include <hip/hip_runtime.h>
#include <hip/hip_bf16.h>
#include <math.h>

// Problem constants (from reference): B=8, S=1024 -> T=8192 tokens
#define TOK 8192
#define DD  768
#define EE  8
#define FF  3072

typedef __attribute__((ext_vector_type(16))) __bf16 v16bf;
typedef __attribute__((ext_vector_type(8)))  __bf16 v8bf;
typedef __attribute__((ext_vector_type(8)))  float  v8f;

union BF16x16 { v16bf v; v8bf h[2]; };

// ---------------------------------------------------------------- init counts
__global__ __launch_bounds__(32) void init_cnt_kernel(int* cnt) {
  if (threadIdx.x < EE) cnt[threadIdx.x] = 0;
}

// ---------------------------------------------------------------- gating
// One wave (32 lanes) per token: partial dots over D, butterfly reduce,
// argmax of logits (== argmax of softmax).
__global__ __launch_bounds__(128) void gate_kernel(const float* __restrict__ x,
                                                   const float* __restrict__ gw,
                                                   const float* __restrict__ gb,
                                                   int* __restrict__ eidx) {
  int wv   = threadIdx.x >> 5;
  int lane = threadIdx.x & 31;
  int t = blockIdx.x * 4 + wv;
  if (t >= TOK) return;
  float acc[EE];
#pragma unroll
  for (int e = 0; e < EE; ++e) acc[e] = 0.f;
  const float* xr = x + (size_t)t * DD;
  for (int k = lane; k < DD; k += 32) {
    float xv = xr[k];
    const float* g = gw + (size_t)k * EE;
#pragma unroll
    for (int e = 0; e < EE; ++e) acc[e] += xv * g[e];
  }
#pragma unroll
  for (int e = 0; e < EE; ++e) {
#pragma unroll
    for (int m = 16; m >= 1; m >>= 1) acc[e] += __shfl_xor(acc[e], m, 32);
  }
  if (lane == 0) {
    int best = 0; float bv = acc[0] + gb[0];
#pragma unroll
    for (int e = 1; e < EE; ++e) {
      float v = acc[e] + gb[e];
      if (v > bv) { bv = v; best = e; }
    }
    eidx[t] = best;
  }
}

// ---------------------------------------------------------------- routing
__global__ __launch_bounds__(256) void scatter_kernel(const int* __restrict__ eidx,
                                                      int* __restrict__ cnt,
                                                      int* __restrict__ bucket) {
  int t = blockIdx.x * 256 + threadIdx.x;
  if (t >= TOK) return;
  int e = eidx[t];
  int pos = atomicAdd(&cnt[e], 1);
  bucket[e * TOK + pos] = t;
}

// ------------------------------------------------- weight transpose + fp32->bf16
// in: per-expert row-major [R,C] fp32 ; out: per-expert row-major [C,R] bf16
__global__ __launch_bounds__(256) void transpose_cvt_kernel(const float* __restrict__ in,
                                                            unsigned short* __restrict__ outp,
                                                            int R, int C) {
  __shared__ float tile[32][33];
  __bf16* out = (__bf16*)outp;
  int e = blockIdx.z;
  const float* ip = in + (size_t)e * R * C;
  __bf16*      op = out + (size_t)e * R * C;
  int c0 = blockIdx.x * 32, r0 = blockIdx.y * 32;
  int tx = threadIdx.x & 31, ty = threadIdx.x >> 5;   // 32 x 8
#pragma unroll
  for (int i = 0; i < 4; ++i)
    tile[ty + 8 * i][tx] = ip[(size_t)(r0 + ty + 8 * i) * C + (c0 + tx)];
  __syncthreads();
#pragma unroll
  for (int i = 0; i < 4; ++i)
    op[(size_t)(c0 + ty + 8 * i) * R + (r0 + tx)] = (__bf16)tile[tx][ty + 8 * i];
}

// ---------------------------------------------------------------- fused MoE FFN
// Per block: 32 tokens (two 16-row m-fragments) of one expert. 8 wave32.
// Stream over F in chunks of 128: h = gelu(x@W1 chunk), then y += h @ W2 chunk.
// Register blocking: each wave computes BOTH m-frags, so B-fragments are loaded
// once and used twice; stage-B A-fragments are loaded once and used 6x.
#define XP 776   // xs row stride (bf16 elems), mult of 8 -> 16B aligned b128 loads
#define HP 136   // hs row stride
#define NC 128   // F chunk (8 waves x 16 h-columns)

__global__ __launch_bounds__(256) void moe_ffn_kernel(
    const float* __restrict__ x,
    const unsigned short* __restrict__ w1tp,   // [E][F][D] bf16 (W1 transposed)
    const float* __restrict__ b1,              // [E][F]
    const unsigned short* __restrict__ w2tp,   // [E][D][F] bf16 (W2 transposed)
    const float* __restrict__ b2,              // [E][D]
    const int* __restrict__ cnt,
    const int* __restrict__ bucket,
    float* __restrict__ out) {
  const __bf16* w1t = (const __bf16*)w1tp;
  const __bf16* w2t = (const __bf16*)w2tp;

  int e = blockIdx.y;
  int mbase = blockIdx.x * 32;
  int ce = cnt[e];
  if (mbase >= ce) return;
  int valid = ce - mbase; if (valid > 32) valid = 32;

  __shared__ int    toks[32];
  __shared__ __bf16 xs[32 * XP];
  __shared__ __bf16 hs[32 * HP];

  if (threadIdx.x < 32) {
    int slot = mbase + threadIdx.x;
    if (slot >= ce) slot = ce - 1;          // pad rows replicate a valid token
    toks[threadIdx.x] = bucket[e * TOK + slot];
  }
  __syncthreads();
  // Gather 32 token rows, convert fp32 -> bf16 into LDS
  for (int idx = threadIdx.x; idx < 32 * DD; idx += 256) {
    int r = idx / DD, c = idx - r * DD;
    xs[r * XP + c] = (__bf16)x[(size_t)toks[r] * DD + c];
  }
  __syncthreads();

  int wv   = threadIdx.x >> 5;   // wave 0..7
  int lane = threadIdx.x & 31;
  int hl   = lane >> 4;          // lane half (A: K-half select, C: M-half select)
  int rm   = lane & 15;          // row (A) / column (B,C) within fragment

  v8f yacc[12] = {};   // [nf][mi]: this wave's 96 y-columns x 2 m-frags (96 VGPRs)

  for (int fc = 0; fc < FF; fc += NC) {
    // ---- Stage A: h_chunk[32, NC], wave wv owns columns fc+wv*16 .. +15
    v8f hacc0 = {}, hacc1 = {};
    const __bf16* w1row = w1t + ((size_t)e * FF + fc + wv * 16 + rm) * DD;
    if (fc + NC < FF) __builtin_prefetch(w1row + (size_t)NC * DD, 0, 1);
#pragma unroll 4
    for (int k0 = 0; k0 < DD; k0 += 32) {
      BF16x16 a0, a1, b;
      // B 32x16 bf16: lane<16 col=rm K{k0..k0+7, k0+16..23}; lane>=16 K{+8,+24}
      const __bf16* bp = w1row + k0 + hl * 8;
      b.h[0] = *(const v8bf*)bp;
      b.h[1] = *(const v8bf*)(bp + 16);
      // A 16x32 bf16: lane<16 row=rm K{k0..k0+7, k0+16..23}; lane>=16 K{+8,+24}
      const __bf16* ap0 = &xs[rm * XP + k0 + hl * 8];
      a0.h[0] = *(const v8bf*)ap0;
      a0.h[1] = *(const v8bf*)(ap0 + 16);
      const __bf16* ap1 = &xs[(16 + rm) * XP + k0 + hl * 8];
      a1.h[0] = *(const v8bf*)ap1;
      a1.h[1] = *(const v8bf*)(ap1 + 16);
      hacc0 = __builtin_amdgcn_wmma_f32_16x16x32_bf16(false, a0.v, false, b.v,
                                                      (short)0, hacc0, false, false);
      hacc1 = __builtin_amdgcn_wmma_f32_16x16x32_bf16(false, a1.v, false, b.v,
                                                      (short)0, hacc1, false, false);
    }
    float bias1 = b1[(size_t)e * FF + fc + wv * 16 + rm];
#pragma unroll
    for (int i = 0; i < 8; ++i) {
      float v0 = hacc0[i] + bias1;
      v0 = 0.5f * v0 * (1.0f + erff(v0 * 0.70710678118654752f));   // exact gelu
      hs[(i + hl * 8) * HP + wv * 16 + rm] = (__bf16)v0;
      float v1 = hacc1[i] + bias1;
      v1 = 0.5f * v1 * (1.0f + erff(v1 * 0.70710678118654752f));
      hs[(16 + i + hl * 8) * HP + wv * 16 + rm] = (__bf16)v1;
    }
    __syncthreads();

    // ---- Stage B: y[:, wv*96 .. +95] += h_chunk @ W2[fc..fc+NC, cols]
#pragma unroll
    for (int kk = 0; kk < NC; kk += 32) {
      BF16x16 a0, a1;
      const __bf16* ap0 = &hs[rm * HP + kk + hl * 8];
      a0.h[0] = *(const v8bf*)ap0;
      a0.h[1] = *(const v8bf*)(ap0 + 16);
      const __bf16* ap1 = &hs[(16 + rm) * HP + kk + hl * 8];
      a1.h[0] = *(const v8bf*)ap1;
      a1.h[1] = *(const v8bf*)(ap1 + 16);
#pragma unroll
      for (int nf = 0; nf < 6; ++nf) {
        int ncol = wv * 96 + nf * 16 + rm;
        const __bf16* bp = w2t + ((size_t)e * DD + ncol) * FF + fc + kk + hl * 8;
        if (kk == 0 && fc + NC < FF) __builtin_prefetch(bp + NC, 0, 1);
        BF16x16 b;
        b.h[0] = *(const v8bf*)bp;
        b.h[1] = *(const v8bf*)(bp + 16);
        yacc[nf * 2 + 0] = __builtin_amdgcn_wmma_f32_16x16x32_bf16(
            false, a0.v, false, b.v, (short)0, yacc[nf * 2 + 0], false, false);
        yacc[nf * 2 + 1] = __builtin_amdgcn_wmma_f32_16x16x32_bf16(
            false, a1.v, false, b.v, (short)0, yacc[nf * 2 + 1], false, false);
      }
    }
    __syncthreads();   // hs reused next chunk
  }

  // ---- Epilogue: +b2, scatter rows back to their tokens
#pragma unroll
  for (int nf = 0; nf < 6; ++nf) {
    int ncol = wv * 96 + nf * 16 + rm;
    float bias2 = b2[(size_t)e * DD + ncol];
#pragma unroll
    for (int mi = 0; mi < 2; ++mi) {
#pragma unroll
      for (int i = 0; i < 8; ++i) {
        int m = mi * 16 + i + hl * 8;
        if (m < valid) out[(size_t)toks[m] * DD + ncol] = yacc[nf * 2 + mi][i] + bias2;
      }
    }
  }
}

// ---------------------------------------------------------------- launch
extern "C" void kernel_launch(void* const* d_in, const int* in_sizes, int n_in,
                              void* d_out, int out_size, void* d_ws, size_t ws_size,
                              hipStream_t stream) {
  const float* x      = (const float*)d_in[0];   // [B,S,D]
  const float* gate_w = (const float*)d_in[1];   // [D,E]
  const float* gate_b = (const float*)d_in[2];   // [E]
  const float* w1     = (const float*)d_in[3];   // [E,D,F]
  const float* b1     = (const float*)d_in[4];   // [E,F]
  const float* w2     = (const float*)d_in[5];   // [E,F,D]
  const float* b2     = (const float*)d_in[6];   // [E,D]
  float* out = (float*)d_out;

  char* ws = (char*)d_ws;
  size_t off = 0;
  int* eidx   = (int*)(ws + off); off += (size_t)TOK * 4;
  int* cnt    = (int*)(ws + off); off += 256;
  int* bucket = (int*)(ws + off); off += (size_t)EE * TOK * 4;
  off = (off + 255) & ~(size_t)255;
  unsigned short* w1t = (unsigned short*)(ws + off); off += (size_t)EE * DD * FF * 2;
  unsigned short* w2t = (unsigned short*)(ws + off); off += (size_t)EE * DD * FF * 2;

  hipLaunchKernelGGL(init_cnt_kernel, dim3(1), dim3(32), 0, stream, cnt);
  hipLaunchKernelGGL(gate_kernel, dim3(TOK / 4), dim3(128), 0, stream,
                     x, gate_w, gate_b, eidx);
  hipLaunchKernelGGL(scatter_kernel, dim3(TOK / 256), dim3(256), 0, stream,
                     eidx, cnt, bucket);
  // W1: [D,F] -> w1t [F,D] ; W2: [F,D] -> w2t [D,F]
  hipLaunchKernelGGL(transpose_cvt_kernel, dim3(FF / 32, DD / 32, EE), dim3(256), 0, stream,
                     w1, w1t, DD, FF);
  hipLaunchKernelGGL(transpose_cvt_kernel, dim3(DD / 32, FF / 32, EE), dim3(256), 0, stream,
                     w2, w2t, FF, DD);
  hipLaunchKernelGGL(moe_ffn_kernel, dim3(TOK / 32, EE), dim3(256), 0, stream,
                     x, w1t, b1, w2t, b2, cnt, bucket, out);
}
